// DeltaNet_22488448762175
// MI455X (gfx1250) — matrix operand order repaired and checked by reading, not compiled, not run on hardware
//
#include <hip/hip_runtime.h>
#include <cstdint>
#include <cstddef>

// ---------------------------------------------------------------------------
// DeltaNet forward for MI455X (gfx1250, wave32, WMMA bf16, async LDS copy)
// ---------------------------------------------------------------------------

#define B_ 4
#define S_ 2048
#define H_ 1024
#define NH_ 16
#define DH_ 64
#define NTOK (B_ * S_)          // 8192
#define EPS_ 1e-6f
#define LN_EPS_ 1e-5f
#define BETA_MIN_ 0.9f
#define BETA_MAX_ 0.9995f
#define ESPLIT 2
#define EW (DH_ / ESPLIT)       // 32 value dims per scan block

typedef __attribute__((ext_vector_type(16))) __bf16 v16bf;
typedef __attribute__((ext_vector_type(8)))  float  v8f;

struct __attribute__((aligned(16))) U4 { unsigned x, y, z, w; };

union Frag16 {
    v16bf v;
    unsigned short s[16];
    U4 q[2];
};

__device__ __forceinline__ unsigned short f2bf(float f) {
    union { float f; unsigned u; } c; c.f = f;
    unsigned u = c.u + 0x7fffu + ((c.u >> 16) & 1u);   // round to nearest even
    return (unsigned short)(u >> 16);
}
__device__ __forceinline__ float bf2f(unsigned short h) {
    union { unsigned u; float f; } c; c.u = ((unsigned)h) << 16;
    return c.f;
}
__device__ __forceinline__ float sigf(float x) { return 1.0f / (1.0f + __expf(-x)); }

__device__ __forceinline__ v8f wmma_bf16(v16bf a, v16bf b, v8f c) {
    return __builtin_amdgcn_wmma_f32_16x16x32_bf16(
        /*neg_a=*/false, a, /*neg_b=*/false, b,
        /*c_mod=*/(short)0, c, /*reuse_a=*/false, /*reuse_b=*/false);
}

__device__ __forceinline__ float block_reduce_256(float v, float* red) {
    int t = threadIdx.x;
    red[t] = v; __syncthreads();
    for (int s = 128; s > 0; s >>= 1) {
        if (t < s) red[t] += red[t + s];
        __syncthreads();
    }
    float r = red[0];
    __syncthreads();
    return r;
}

// ---------------------------------------------------------------------------
// 1) fp32 -> bf16 weight conversion
// ---------------------------------------------------------------------------
__global__ void cvt_bf16_kernel(const float* __restrict__ src,
                                unsigned short* __restrict__ dst, int n) {
    int i = blockIdx.x * 256 + threadIdx.x;
    if (i < n) dst[i] = f2bf(src[i]);
}

// ---------------------------------------------------------------------------
// 2) fused RMSNorm: xn = rmsnorm(x, rin); kin = rmsnorm(xn, rk); vin = rmsnorm(xn, rv)
// ---------------------------------------------------------------------------
__global__ __launch_bounds__(256) void rms_kernel(
    const float* __restrict__ x, const float* __restrict__ rin,
    const float* __restrict__ rk, const float* __restrict__ rv,
    unsigned short* __restrict__ xnb, unsigned short* __restrict__ kb,
    unsigned short* __restrict__ vb)
{
    __shared__ float red[256];
    long token = blockIdx.x;
    const float* xr = x + token * H_;
    int t = threadIdx.x;

    float xv[4]; float ss = 0.f;
    #pragma unroll
    for (int j = 0; j < 4; j++) { xv[j] = xr[t + 256 * j]; ss += xv[j] * xv[j]; }
    float tot = block_reduce_256(ss, red);
    float rms1 = rsqrtf(tot * (1.0f / H_) + EPS_);

    float xn[4]; float ss2 = 0.f;
    #pragma unroll
    for (int j = 0; j < 4; j++) {
        xn[j] = xv[j] * rms1 * rin[t + 256 * j];
        ss2 += xn[j] * xn[j];
    }
    float tot2 = block_reduce_256(ss2, red);
    float rms2 = rsqrtf(tot2 * (1.0f / H_) + EPS_);

    #pragma unroll
    for (int j = 0; j < 4; j++) {
        int i = t + 256 * j;
        xnb[token * H_ + i] = f2bf(xn[j]);
        kb [token * H_ + i] = f2bf(xn[j] * rms2 * rk[i]);
        vb [token * H_ + i] = f2bf(xn[j] * rms2 * rv[i]);
    }
}

// ---------------------------------------------------------------------------
// 3) beta projection
// ---------------------------------------------------------------------------
__global__ __launch_bounds__(128) void beta_kernel(
    const unsigned short* __restrict__ xnb, const float* __restrict__ bw,
    const float* __restrict__ bb, const float* __restrict__ base_logit,
    float* __restrict__ beta)
{
    long token = blockIdx.x;
    int lane = threadIdx.x & 31, wave = threadIdx.x >> 5;
    const unsigned short* xr = xnb + token * H_;
    for (int h = wave * 4; h < wave * 4 + 4; h++) {
        const float* wr = bw + (long)h * H_;
        float s = 0.f;
        for (int i = lane; i < H_; i += 32) s += bf2f(xr[i]) * wr[i];
        #pragma unroll
        for (int m = 16; m >= 1; m >>= 1) s += __shfl_xor(s, m, 32);
        if (lane == 0) {
            float bv = sigf(s + bb[h]) * sigf(base_logit[h]);
            bv = fminf(BETA_MAX_, fmaxf(BETA_MIN_, bv));
            beta[token * NH_ + h] = bv;
        }
    }
}

// ---------------------------------------------------------------------------
// 4) bf16 WMMA GEMM:  C[M,N] = A[M,K](bf16) * W[N,K]^T(bf16)  (+bias)
//    Block: 256 thr = 8 waves; block tile 16 x 512; wave tile 16 x 64.
//    A tile staged in LDS via GLOBAL_LOAD_ASYNC_TO_LDS_B128 (ASYNCcnt path).
// ---------------------------------------------------------------------------
__global__ __launch_bounds__(256) void gemm_bf16_kernel(
    const unsigned short* __restrict__ A, const unsigned short* __restrict__ W,
    const float* __restrict__ bias, float* __restrict__ Cf,
    unsigned short* __restrict__ Cb, int M, int N, int K)
{
    __shared__ __attribute__((aligned(16))) unsigned short At[16 * 32];

    int mbase = blockIdx.x * 16;
    int nblock = blockIdx.y * 512;
    int t = threadIdx.x, lane = t & 31, wave = t >> 5;
    int nbase = nblock + wave * 64;
    int hi = lane >> 4, lo = lane & 15;

    v8f acc[4] = {};
    for (int k0 = 0; k0 < K; k0 += 32) {
        // async copy of the 16x32 bf16 A tile (1 KB) straight into LDS:
        // 64 lanes each move one 16-byte chunk (row = t>>2, chunk = t&3)
        if (t < 64) {
            int row = t >> 2, ch = t & 3;
            const unsigned short* src = A + (long)(mbase + row) * K + k0 + ch * 8;
            unsigned lds_off = (unsigned)(size_t)&At[row * 32 + ch * 8];
            asm volatile("global_load_async_to_lds_b128 %0, %1, off"
                         :: "v"(lds_off), "v"((unsigned long long)(size_t)src)
                         : "memory");
        }
        asm volatile("s_wait_asynccnt 0x0" ::: "memory");
        __syncthreads();

        Frag16 fa;
        fa.q[0] = *(const U4*)&At[lo * 32 + hi * 8];
        fa.q[1] = *(const U4*)&At[lo * 32 + 16 + hi * 8];

        #pragma unroll
        for (int sub = 0; sub < 4; sub++) {
            const unsigned short* wr =
                W + (long)(nbase + sub * 16 + lo) * K + k0 + hi * 8;
            __builtin_prefetch(wr + 32, 0, 1);      // next k-step of this row
            Frag16 fb;
            fb.q[0] = *(const U4*)(wr);
            fb.q[1] = *(const U4*)(wr + 16);
            acc[sub] = wmma_bf16(fa.v, fb.v, acc[sub]);
        }
        __syncthreads();
    }

    #pragma unroll
    for (int sub = 0; sub < 4; sub++) {
        int nn = nbase + sub * 16 + lo;
        float bv = bias ? bias[nn] : 0.f;
        #pragma unroll
        for (int j = 0; j < 8; j++) {
            int mm = mbase + hi * 8 + j;
            float v = acc[sub][j] + bv;
            long off = (long)mm * N + nn;
            if (Cf) Cf[off] = v;
            if (Cb) Cb[off] = f2bf(v);
        }
    }
}

// ---------------------------------------------------------------------------
// 5) chunked DeltaNet scan. Grid = B*NH*ESPLIT blocks; each block owns a
//    64 x EW slice of the state (value dims are independent). Chunk=64 tokens.
//      P_i = prefix prod of beta; den_i = beta*den + 1
//      S   = Q K^T, weighted w_ij = (j<=i) ? P_i/P_j : 0   (shared across e)
//      O   = Sw * V  +  (diag(P) Q) * M0
//      dM  = (diag(P_C/P_j) K)^T V ;  M <- P_C*M + dM
// ---------------------------------------------------------------------------
__global__ __launch_bounds__(256) void scan_kernel(
    const unsigned short* __restrict__ qg, const unsigned short* __restrict__ kg,
    const unsigned short* __restrict__ vg, const float* __restrict__ beta,
    unsigned short* __restrict__ outg)
{
    __shared__ float Ms[64 * EW];                                        // M[d][el] f32
    __shared__ __attribute__((aligned(16))) unsigned short Mt [EW * 64]; // M^T[el][d] bf16
    __shared__ __attribute__((aligned(16))) unsigned short Vt [EW * 64]; // V^T[el][j]
    __shared__ __attribute__((aligned(16))) unsigned short Kwt[64 * 64]; // (wK)^T[d][j]
    __shared__ __attribute__((aligned(16))) unsigned short Sw [64 * 64]; // Sw[i][j]
    __shared__ float Pp[64], Dn[64];
    __shared__ float dcarry;

    const int blk = blockIdx.x;          // 0 .. B*NH*ESPLIT-1
    const int bh = blk >> 1, eh = blk & 1;
    const int b = bh >> 4, h = bh & 15;
    const int e0blk = eh * EW;
    const int t = threadIdx.x, lane = t & 31, wave = t >> 5;
    const int hi = lane >> 4, lo = lane & 15;

    for (int i = t; i < 64 * EW; i += 256) { Ms[i] = 0.f; Mt[i] = 0; }
    if (t == 0) dcarry = 0.f;
    __syncthreads();

    for (int c = 0; c < S_ / 64; c++) {
        const long tok0 = (long)b * S_ + c * 64;

        // stage V^T slice into LDS: Vt[el][j] = v[j][e0blk+el]
        for (int idx = t; idx < 64 * EW; idx += 256) {
            int j = idx >> 5, el = idx & (EW - 1);
            Vt[el * 64 + j] = vg[(tok0 + j) * H_ + h * DH_ + e0blk + el];
        }
        // prefix decay products and denominator (tiny serial part)
        if (t == 0) {
            float p = 1.f, d = dcarry;
            for (int i = 0; i < 64; i++) {
                float bt = beta[(tok0 + i) * NH_ + h];
                p *= bt; d = d * bt + 1.f;
                Pp[i] = p; Dn[i] = d;
            }
            dcarry = d;
        }
        __syncthreads();

        const float PC = Pp[63];
        // Kwt[d][j] = (PC/P_j) * k_j[d]
        for (int idx = t; idx < 4096; idx += 256) {
            int j = idx >> 6, d = idx & 63;
            float kv = bf2f(kg[(tok0 + j) * H_ + h * DH_ + d]);
            Kwt[d * 64 + j] = f2bf(kv * (PC / Pp[j]));
        }
        __syncthreads();

        // ---- GEMM 1: S = Q K^T, decay-mask, -> Sw (bf16); 16 tiles, 2/wave ----
        for (int tt = wave * 2; tt < wave * 2 + 2; tt++) {
            int i0 = (tt >> 2) * 16, j0 = (tt & 3) * 16;
            v8f acc = {};
            const unsigned short* arow = qg + (tok0 + i0 + lo) * H_ + h * DH_;
            const unsigned short* brow = kg + (tok0 + j0 + lo) * H_ + h * DH_;
            #pragma unroll
            for (int k0 = 0; k0 < 64; k0 += 32) {
                Frag16 fa, fb;
                fa.q[0] = *(const U4*)(arow + k0 + hi * 8);
                fa.q[1] = *(const U4*)(arow + k0 + 16 + hi * 8);
                fb.q[0] = *(const U4*)(brow + k0 + hi * 8);
                fb.q[1] = *(const U4*)(brow + k0 + 16 + hi * 8);
                acc = wmma_bf16(fa.v, fb.v, acc);
            }
            int nn = j0 + lo;
            #pragma unroll
            for (int j = 0; j < 8; j++) {
                int mm = i0 + hi * 8 + j;
                float w = (nn <= mm) ? (Pp[mm] / Pp[nn]) : 0.f;
                Sw[mm * 64 + nn] = f2bf(acc[j] * w);
            }
        }
        __syncthreads();

        // ---- GEMM 2+3: O = Sw*V + (diag(P) Q)*M0 ; 8 tiles, 1/wave ----
        {
            int i0 = (wave >> 1) * 16, e0l = (wave & 1) * 16;
            v8f acc = {};
            const unsigned short* arow = &Sw[(i0 + lo) * 64];
            const unsigned short* brow = &Vt[(e0l + lo) * 64];
            #pragma unroll
            for (int k0 = 0; k0 < 64; k0 += 32) {
                Frag16 fa, fb;
                fa.q[0] = *(const U4*)(arow + k0 + hi * 8);
                fa.q[1] = *(const U4*)(arow + k0 + 16 + hi * 8);
                fb.q[0] = *(const U4*)(brow + k0 + hi * 8);
                fb.q[1] = *(const U4*)(brow + k0 + 16 + hi * 8);
                acc = wmma_bf16(fa.v, fb.v, acc);
            }
            const unsigned short* qrow = qg + (tok0 + i0 + lo) * H_ + h * DH_;
            const unsigned short* mrow = &Mt[(e0l + lo) * 64];
            float ps = Pp[i0 + lo];
            #pragma unroll
            for (int k0 = 0; k0 < 64; k0 += 32) {
                Frag16 fa, fb;
                #pragma unroll
                for (int e2 = 0; e2 < 8; e2++) {
                    fa.s[e2]     = f2bf(ps * bf2f(qrow[k0 + hi * 8 + e2]));
                    fa.s[8 + e2] = f2bf(ps * bf2f(qrow[k0 + 16 + hi * 8 + e2]));
                }
                fb.q[0] = *(const U4*)(mrow + k0 + hi * 8);
                fb.q[1] = *(const U4*)(mrow + k0 + 16 + hi * 8);
                acc = wmma_bf16(fa.v, fb.v, acc);
            }
            int ee = e0blk + e0l + lo;
            #pragma unroll
            for (int j = 0; j < 8; j++) {
                int ii = i0 + hi * 8 + j;
                float o = acc[j] / (Dn[ii] + EPS_);
                outg[(tok0 + ii) * H_ + h * DH_ + ee] = f2bf(o);
            }
        }
        __syncthreads();

        // ---- GEMM 4: dM = Kw^T V ; M <- PC*M + dM ; 8 tiles, 1/wave ----
        {
            int d0 = (wave >> 1) * 16, e0l = (wave & 1) * 16;
            v8f acc = {};
            const unsigned short* arow = &Kwt[(d0 + lo) * 64];
            const unsigned short* brow = &Vt [(e0l + lo) * 64];
            #pragma unroll
            for (int k0 = 0; k0 < 64; k0 += 32) {
                Frag16 fa, fb;
                fa.q[0] = *(const U4*)(arow + k0 + hi * 8);
                fa.q[1] = *(const U4*)(arow + k0 + 16 + hi * 8);
                fb.q[0] = *(const U4*)(brow + k0 + hi * 8);
                fb.q[1] = *(const U4*)(brow + k0 + 16 + hi * 8);
                acc = wmma_bf16(fa.v, fb.v, acc);
            }
            int eel = e0l + lo;
            #pragma unroll
            for (int j = 0; j < 8; j++) {
                int dd = d0 + hi * 8 + j;
                float nv = PC * Ms[dd * EW + eel] + acc[j];
                Ms[dd * EW + eel] = nv;
                Mt[eel * 64 + dd] = f2bf(nv);
            }
        }
        __syncthreads();
    }
}

// ---------------------------------------------------------------------------
// 6) residual + LayerNorm
// ---------------------------------------------------------------------------
__global__ __launch_bounds__(256) void ln_kernel(
    const float* __restrict__ x, const float* __restrict__ proj,
    const float* __restrict__ lnw, const float* __restrict__ lnb,
    float* __restrict__ out)
{
    __shared__ float red[256];
    long token = blockIdx.x;
    const float* xr = x + token * H_;
    const float* pr = proj + token * H_;
    int t = threadIdx.x;

    float y[4]; float s = 0.f;
    #pragma unroll
    for (int j = 0; j < 4; j++) { y[j] = xr[t + 256 * j] + pr[t + 256 * j]; s += y[j]; }
    float mu = block_reduce_256(s, red) * (1.0f / H_);
    float vs = 0.f;
    #pragma unroll
    for (int j = 0; j < 4; j++) { float d = y[j] - mu; vs += d * d; }
    float var = block_reduce_256(vs, red) * (1.0f / H_);
    float inv = rsqrtf(var + LN_EPS_);
    #pragma unroll
    for (int j = 0; j < 4; j++) {
        int i = t + 256 * j;
        out[token * H_ + i] = (y[j] - mu) * inv * lnw[i] + lnb[i];
    }
}

// ---------------------------------------------------------------------------
// launcher
// ---------------------------------------------------------------------------
extern "C" void kernel_launch(void* const* d_in, const int* in_sizes, int n_in,
                              void* d_out, int out_size, void* d_ws, size_t ws_size,
                              hipStream_t stream)
{
    (void)in_sizes; (void)n_in; (void)out_size; (void)ws_size;

    const float* x          = (const float*)d_in[0];
    const float* rin_w      = (const float*)d_in[1];
    const float* rk_w       = (const float*)d_in[2];
    const float* rv_w       = (const float*)d_in[3];
    const float* Wq         = (const float*)d_in[4];
    const float* Wk         = (const float*)d_in[5];
    const float* Wv         = (const float*)d_in[6];
    const float* beta_w     = (const float*)d_in[7];
    const float* beta_b     = (const float*)d_in[8];
    const float* base_logit = (const float*)d_in[9];
    const float* Wo         = (const float*)d_in[10];
    const float* bo         = (const float*)d_in[11];
    const float* ln_w       = (const float*)d_in[12];
    const float* ln_b       = (const float*)d_in[13];
    float* out = (float*)d_out;

    char* ws = (char*)d_ws;
    const size_t TOKH2 = (size_t)NTOK * H_ * 2;          // 16 MB (bf16 [8192,1024])
    const size_t W2    = (size_t)H_ * H_ * 2;            // 2 MB  (bf16 weight)

    unsigned short* xn_bf  = (unsigned short*)(ws);                      // +0
    unsigned short* kin_bf = (unsigned short*)(ws + TOKH2);              // +16M
    unsigned short* vin_bf = (unsigned short*)(ws + 2 * TOKH2);          // +32M
    unsigned short* Wq_bf  = (unsigned short*)(ws + 3 * TOKH2);          // +48M
    unsigned short* Wk_bf  = (unsigned short*)(ws + 3 * TOKH2 + W2);
    unsigned short* Wv_bf  = (unsigned short*)(ws + 3 * TOKH2 + 2 * W2);
    unsigned short* Wo_bf  = (unsigned short*)(ws + 3 * TOKH2 + 3 * W2);
    unsigned short* q_bf   = (unsigned short*)(ws + 3 * TOKH2 + 4 * W2);          // +56M
    unsigned short* k_bf   = (unsigned short*)(ws + 4 * TOKH2 + 4 * W2);          // +72M
    unsigned short* v_bf   = (unsigned short*)(ws + 5 * TOKH2 + 4 * W2);          // +88M
    float*          beta   = (float*)         (ws + 6 * TOKH2 + 4 * W2);          // +104M
    // aliases (consumed-before-reuse, stream-ordered):
    float*          proj   = (float*)xn_bf;            // 32 MB over xn_bf+kin_bf
    unsigned short* out_bf = vin_bf;                   // 16 MB, after v GEMM done

    const int WN = H_ * H_;                            // 1048576
    dim3 cvtg((WN + 255) / 256);
    cvt_bf16_kernel<<<cvtg, 256, 0, stream>>>(Wq, Wq_bf, WN);
    cvt_bf16_kernel<<<cvtg, 256, 0, stream>>>(Wk, Wk_bf, WN);
    cvt_bf16_kernel<<<cvtg, 256, 0, stream>>>(Wv, Wv_bf, WN);
    cvt_bf16_kernel<<<cvtg, 256, 0, stream>>>(Wo, Wo_bf, WN);

    rms_kernel<<<NTOK, 256, 0, stream>>>(x, rin_w, rk_w, rv_w, xn_bf, kin_bf, vin_bf);
    beta_kernel<<<NTOK, 128, 0, stream>>>(xn_bf, beta_w, beta_b, base_logit, beta);

    dim3 gg(NTOK / 16, H_ / 512);                      // (512, 2)
    gemm_bf16_kernel<<<gg, 256, 0, stream>>>(xn_bf,  Wq_bf, nullptr, nullptr, q_bf, NTOK, H_, H_);
    gemm_bf16_kernel<<<gg, 256, 0, stream>>>(kin_bf, Wk_bf, nullptr, nullptr, k_bf, NTOK, H_, H_);
    gemm_bf16_kernel<<<gg, 256, 0, stream>>>(vin_bf, Wv_bf, nullptr, nullptr, v_bf, NTOK, H_, H_);

    scan_kernel<<<B_ * NH_ * ESPLIT, 256, 0, stream>>>(q_bf, k_bf, v_bf, beta, out_bf);

    gemm_bf16_kernel<<<gg, 256, 0, stream>>>(out_bf, Wo_bf, bo, proj, nullptr, NTOK, H_, H_);

    ln_kernel<<<NTOK, 256, 0, stream>>>(x, proj, ln_w, ln_b, out);
}
